// DotProductAttention_40742059770510
// MI455X (gfx1250) — matrix operand order repaired and checked by reading, not compiled, not run on hardware
//
#include <hip/hip_runtime.h>
#include <math.h>

typedef __attribute__((ext_vector_type(16))) _Float16 v16h;
typedef __attribute__((ext_vector_type(8)))  _Float16 v8h;
typedef __attribute__((ext_vector_type(4)))  _Float16 v4h;
typedef __attribute__((ext_vector_type(2)))  _Float16 h2;
typedef __attribute__((ext_vector_type(8)))  float    v8f;
typedef __attribute__((ext_vector_type(4)))  float    v4f;
typedef unsigned int u32x4 __attribute__((ext_vector_type(4)));
typedef int          i32x4 __attribute__((ext_vector_type(4)));
typedef int          i32x8 __attribute__((ext_vector_type(8)));

namespace {
constexpr int kB  = 4;
constexpr int kLq = 4096;
constexpr int kLk = 4096;
constexpr int kD  = 64;    // head dim
constexpr int kBQ = 64;    // Q rows per workgroup
constexpr int kBK = 64;    // keys per iteration
constexpr int kNW = 4;     // waves per workgroup (wave32)
constexpr int kThreads = kNW * 32;
// LDS row stride: 64 + 8 pad halves = 144B rows (16B-aligned runs, bank-spread).
// TDM reproduces this via pad_interval=32 dwords, pad_amount=4 dwords.
constexpr int STR = 72;
constexpr float kQScale = 0.125f * 1.44269504088896340736f;  // log2(e)/sqrt(64)
}

#if defined(__has_builtin)
#if __has_builtin(__builtin_amdgcn_tensor_load_to_lds) && \
    __has_builtin(__builtin_amdgcn_s_wait_tensorcnt)
#define HAVE_TDM 1
#endif
#endif

__device__ __forceinline__ v16h ld16(const _Float16* p) {
  // B-fragment: 16 contiguous halves (lane = column, k = hi*16..hi*16+15)
  v8h a = *(const v8h*)(p);
  v8h b = *(const v8h*)(p + 8);
  return __builtin_shufflevector(a, b, 0,1,2,3,4,5,6,7,8,9,10,11,12,13,14,15);
}

__device__ __forceinline__ v16h ld8x2(const _Float16* p) {
  // A-fragment: two 8-half runs at p and p+16 (K = base..+7, base+16..+23)
  v8h a = *(const v8h*)(p);
  v8h b = *(const v8h*)(p + 16);
  return __builtin_shufflevector(a, b, 0,1,2,3,4,5,6,7,8,9,10,11,12,13,14,15);
}

__device__ __forceinline__ v8f wmma16(v16h a, v16h b, v8f c) {
  return __builtin_amdgcn_wmma_f32_16x16x32_f16(false, a, false, b, (short)0, c,
                                                false, false);
}

#ifdef HAVE_TDM
// ---- Tensor Data Mover: async copy of a 64x64 f16 2D tile into LDS ----
// D# per CDNA5 ISA ch.8: group0 = {count/type, lds_addr, global_addr[56:0]},
// group1 = {data_size=2B, pad(4dw every 32dw), dims, tile dims, stride}.
__device__ __forceinline__ void tdm_load_2d(unsigned lds_off, unsigned long long gaddr,
                                            unsigned d0, unsigned d1,
                                            unsigned long long stride0) {
  u32x4 g0;
  g0[0] = 1u;                                           // count=1, no gather
  g0[1] = lds_off;                                      // LDS byte address
  g0[2] = (unsigned)gaddr;                              // global addr [31:0]
  g0[3] = (unsigned)((gaddr >> 32) & 0x1FFFFFFu) | (2u << 30);  // [56:32] | type=2
  i32x8 g1;
  g1[0] = (1 << 16) | (1 << 20) | (4 << 22) | (3 << 25);  // dsz=2B, pad_en, 32dw/4dw
  g1[1] = (int)((d0 & 0xFFFFu) << 16);                    // tensor_dim0 [15:0]
  g1[2] = (int)(((d0 >> 16) & 0xFFFFu) | ((d1 & 0xFFFFu) << 16));
  g1[3] = (int)(((d1 >> 16) & 0xFFFFu) | (64u << 16));    // tile_dim0 = 64
  g1[4] = 64;                                             // tile_dim1=64, tile_dim2=0
  g1[5] = (int)(unsigned)stride0;                         // dim0 stride [31:0]
  g1[6] = (int)(unsigned)((stride0 >> 32) & 0xFFFFu);     // stride [47:32]; dim1_stride=0
  g1[7] = 0;
  i32x4 gz = {0, 0, 0, 0};
#if defined(__clang_major__) && __clang_major__ >= 23
  i32x8 gz8 = {0, 0, 0, 0, 0, 0, 0, 0};
  __builtin_amdgcn_tensor_load_to_lds(g0, g1, gz, gz, gz8, 0);
#else
  __builtin_amdgcn_tensor_load_to_lds(g0, g1, gz, gz, 0);
#endif
}
#endif  // HAVE_TDM

// ======== pre-pass 1: Q (scaled) and K -> f16, mask -> f32 bias ========
__global__ __launch_bounds__(256) void prep_cvt(
    const float* __restrict__ Q, const float* __restrict__ K,
    const int* __restrict__ M, _Float16* __restrict__ Qh,
    _Float16* __restrict__ Kh, float* __restrict__ Bw) {
  const int nq4 = kB * kLq * kD / 4;       // 262144
  const int nk4 = kB * kLk * kD / 4;       // 262144
  const int nm  = kB * kLk;                // 16384
  int i = blockIdx.x * 256 + threadIdx.x;
  if (i < nq4) {
    v4f d = ((const v4f*)Q)[i];
    v4h h;
#pragma unroll
    for (int j = 0; j < 4; ++j) h[j] = (_Float16)(d[j] * kQScale);
    ((v4h*)Qh)[i] = h;
    return;
  }
  int ik = i - nq4;
  if (ik < nk4) {
    v4f d = ((const v4f*)K)[ik];
    v4h h;
#pragma unroll
    for (int j = 0; j < 4; ++j) h[j] = (_Float16)d[j];
    ((v4h*)Kh)[ik] = h;
    return;
  }
  int im = ik - nk4;
  if (im < nm) Bw[im] = M[im] ? 0.0f : -INFINITY;
}

// ======== pre-pass 2: V -> f16 transposed [b][feat][key] ========
__global__ __launch_bounds__(256) void prep_vt(const float* __restrict__ V,
                                               _Float16* __restrict__ Vt) {
  __shared__ float t[64][65];
  const int tid = threadIdx.x;
  const int b   = blockIdx.y;
  const int k0  = blockIdx.x * 64;
  const float* src = V + ((size_t)b * kLk + k0) * kD;
#pragma unroll
  for (int i = 0; i < 4; ++i) {
    int idx = tid + i * 256;
    int row = idx >> 4;
    int c4  = idx & 15;
    v4f d = *(const v4f*)(src + (size_t)row * kD + c4 * 4);
#pragma unroll
    for (int j = 0; j < 4; ++j) t[row][c4 * 4 + j] = d[j];
  }
  __syncthreads();
  const int f = tid >> 2;
  const int g = tid & 3;
  _Float16* dst = Vt + ((size_t)b * kD + f) * kLk + k0 + g * 16;
  v8h h0, h1;
#pragma unroll
  for (int j = 0; j < 8; ++j) h0[j] = (_Float16)t[g * 16 + j][f];
#pragma unroll
  for (int j = 0; j < 8; ++j) h1[j] = (_Float16)t[g * 16 + 8 + j][f];
  *(v8h*)(dst)     = h0;
  *(v8h*)(dst + 8) = h1;
}

// ======== shared compute body (used by both main kernels) ========
struct FaState {
  float m_i[8];
  v8f acc[4];
  v8f lacc;
};

__device__ __forceinline__ void fa_tile(FaState& st, const _Float16* Kt,
                                        const _Float16* Vv, const float* Bias,
                                        _Float16* pw, v16h qa0, v16h qa1,
                                        v16h vone, int ln, int hi) {
  // issue bias reads early; consumed after the S WMMAs
  float bs[4];
#pragma unroll
  for (int f = 0; f < 4; ++f) bs[f] = Bias[f * 16 + ln];

  // ---- S = Q · K^T : batch-load 4 fragments, then 4 independent WMMAs ----
  v8f S[4];
  v16h bk[4];
#pragma unroll
  for (int f = 0; f < 4; ++f) bk[f] = ld8x2(&Kt[(f * 16 + ln) * STR + hi * 8]);
#pragma unroll
  for (int f = 0; f < 4; ++f) {
#pragma unroll
    for (int r = 0; r < 8; ++r) S[f][r] = 0.0f;
    S[f] = wmma16(qa0, bk[f], S[f]);
  }
#pragma unroll
  for (int f = 0; f < 4; ++f) bk[f] = ld8x2(&Kt[(f * 16 + ln) * STR + 32 + hi * 8]);
#pragma unroll
  for (int f = 0; f < 4; ++f) S[f] = wmma16(qa1, bk[f], S[f]);

  // ---- mask bias ----
#pragma unroll
  for (int f = 0; f < 4; ++f) {
#pragma unroll
    for (int r = 0; r < 8; ++r) S[f][r] += bs[f];
  }
  // ---- row max: pairwise f32, cross-lane reduce as packed f16 ----
  float rmax[8];
#pragma unroll
  for (int r = 0; r < 8; ++r)
    rmax[r] = fmaxf(fmaxf(S[0][r], S[1][r]), fmaxf(S[2][r], S[3][r]));
  int pk[4];
#pragma unroll
  for (int rp = 0; rp < 4; ++rp) {
    h2 t = {(_Float16)rmax[2 * rp], (_Float16)rmax[2 * rp + 1]};
    pk[rp] = __builtin_bit_cast(int, t);
  }
#pragma unroll
  for (int off = 8; off >= 1; off >>= 1) {
#pragma unroll
    for (int rp = 0; rp < 4; ++rp) {
      int o = __shfl_xor(pk[rp], off, 32);
      h2 mr = __builtin_elementwise_max(__builtin_bit_cast(h2, pk[rp]),
                                        __builtin_bit_cast(h2, o));
      pk[rp] = __builtin_bit_cast(int, mr);
    }
  }
#pragma unroll
  for (int rp = 0; rp < 4; ++rp) {
    h2 t = __builtin_bit_cast(h2, pk[rp]);
    rmax[2 * rp]     = (float)t[0];
    rmax[2 * rp + 1] = (float)t[1];
  }
  // ---- online softmax update (base-2) ----
  float al[8];
  bool need = false;
#pragma unroll
  for (int r = 0; r < 8; ++r) {
    float nm = fmaxf(st.m_i[r], rmax[r]);
    float a  = (st.m_i[r] == nm) ? 1.0f : exp2f(st.m_i[r] - nm);
    al[r] = a;
    need |= (a != 1.0f);
    st.m_i[r] = nm;
    float mr2 = (nm == -INFINITY) ? 0.0f : nm;   // all-masked-row guard
#pragma unroll
    for (int f = 0; f < 4; ++f) S[f][r] = exp2f(S[f][r] - mr2);
  }
  if (__any((int)need)) {
#pragma unroll
    for (int c = 0; c < 4; ++c)
#pragma unroll
      for (int r = 0; r < 8; ++r) st.acc[c][r] *= al[r];
#pragma unroll
    for (int r = 0; r < 8; ++r) st.lacc[r] *= al[r];
  }
  // ---- P (C-layout) -> per-wave LDS f16 -> A-fragments ----
#pragma unroll
  for (int r = 0; r < 8; ++r)
#pragma unroll
    for (int f = 0; f < 4; ++f)
      pw[(r + 8 * hi) * STR + f * 16 + ln] = (_Float16)S[f][r];
  asm volatile("" ::: "memory");  // same-wave DS is in-order; stop reordering
  v16h pa0 = ld8x2(&pw[ln * STR +  0 + hi * 8]);
  v16h pa1 = ld8x2(&pw[ln * STR + 32 + hi * 8]);

  // ---- O += P·V (8 WMMAs); l += P·1 (2 WMMAs, LDS-independent fillers) ----
  v16h vb[4];
#pragma unroll
  for (int c = 0; c < 4; ++c) vb[c] = ld8x2(&Vv[(c * 16 + ln) * STR + hi * 8]);
  st.lacc = wmma16(pa0, vone, st.lacc);   // runs while the vb loads are in flight
#pragma unroll
  for (int c = 0; c < 4; ++c) st.acc[c] = wmma16(pa0, vb[c], st.acc[c]);
#pragma unroll
  for (int c = 0; c < 4; ++c) vb[c] = ld8x2(&Vv[(c * 16 + ln) * STR + 32 + hi * 8]);
  st.lacc = wmma16(pa1, vone, st.lacc);
#pragma unroll
  for (int c = 0; c < 4; ++c) st.acc[c] = wmma16(pa1, vb[c], st.acc[c]);
}

__device__ __forceinline__ void fa_init(FaState& st) {
#pragma unroll
  for (int r = 0; r < 8; ++r) { st.m_i[r] = -INFINITY; st.lacc[r] = 0.0f; }
#pragma unroll
  for (int c = 0; c < 4; ++c)
#pragma unroll
    for (int r = 0; r < 8; ++r) st.acc[c][r] = 0.0f;
}

__device__ __forceinline__ void fa_epilogue(FaState& st, float* O, int b, int qt,
                                            int wave, int ln, int hi) {
  float linv[8];
#pragma unroll
  for (int r = 0; r < 8; ++r) {
    float lv = __shfl(st.lacc[r], hi << 4, 32);
    linv[r] = 1.0f / lv;
  }
  float* Og = O + ((size_t)b * kLq + qt + wave * 16) * kD;
#pragma unroll
  for (int c = 0; c < 4; ++c)
#pragma unroll
    for (int r = 0; r < 8; ++r) {
      int row = r + 8 * hi;
      Og[(size_t)row * kD + c * 16 + ln] = st.acc[c][r] * linv[r];
    }
}

__device__ __forceinline__ v16h make_vone(int ln) {
  v16h vone;
  _Float16 ov = (ln == 0) ? (_Float16)1.0f : (_Float16)0.0f;
#pragma unroll
  for (int j = 0; j < 16; ++j) vone[j] = ov;
  return vone;
}

#ifdef HAVE_TDM
// ======== main kernel: TDM double-buffered streaming ========
__global__ __launch_bounds__(kThreads) void fa64_tdm(
    const _Float16* __restrict__ Qh, const _Float16* __restrict__ Kh,
    const _Float16* __restrict__ Vt, const float* __restrict__ Bw,
    float* __restrict__ O) {
  __shared__ __align__(16) _Float16 Kbuf[2][kBK * STR];
  __shared__ __align__(16) _Float16 Vbuf[2][kD * STR];
  __shared__ __align__(16) _Float16 Plds[kNW * 16 * STR];
  __shared__ float Bias[kBK];

  const int tid  = threadIdx.x;
  const int wave = tid >> 5;
  const int lane = tid & 31;
  const int ln   = lane & 15;
  const int hi   = lane >> 4;
  const int b    = blockIdx.y;
  const int qt   = blockIdx.x * kBQ;

  const unsigned long long kg =
      (unsigned long long)(uintptr_t)(Kh + (size_t)b * kLk * kD);
  const unsigned long long vg =
      (unsigned long long)(uintptr_t)(Vt + (size_t)b * kD * kLk);
  // LDS byte offsets (flat shared addr truncates to the LDS aperture offset)
  const unsigned klds[2] = {(unsigned)(size_t)(void*)&Kbuf[0][0],
                            (unsigned)(size_t)(void*)&Kbuf[1][0]};
  const unsigned vlds[2] = {(unsigned)(size_t)(void*)&Vbuf[0][0],
                            (unsigned)(size_t)(void*)&Vbuf[1][0]};

  // Q A-fragments straight from global f16 (pre-scaled by prep_cvt)
  const _Float16* Qrow = Qh + ((size_t)b * kLq + qt + wave * 16 + ln) * kD;
  const v16h qa0 = ld8x2(Qrow + hi * 8);
  const v16h qa1 = ld8x2(Qrow + 32 + hi * 8);
  const v16h vone = make_vone(ln);

  FaState st;
  fa_init(st);

  if (wave == 0) {  // prologue: DMA tile 0 into buffer 0
    tdm_load_2d(klds[0], kg, kD, kLk, kD);                 // K: 64x64, stride 64
    tdm_load_2d(vlds[0], vg, kLk, kD, kLk);                // V^T: 64 keys x 64 feats
  }

  for (int kb = 0; kb < kLk; kb += kBK) {
    const int cur = (kb >> 6) & 1;
    if (tid < kBK) Bias[tid] = Bw[(size_t)b * kLk + kb + tid];
    if (wave == 0) {
      if (kb + kBK < kLk) {  // issue next tile, then wait only for the current one
        const int nb = cur ^ 1;
        tdm_load_2d(klds[nb], kg + (size_t)(kb + kBK) * kD * 2, kD, kLk, kD);
        tdm_load_2d(vlds[nb], vg + (size_t)(kb + kBK) * 2, kLk, kD, kLk);
        __builtin_amdgcn_s_wait_tensorcnt((short)2);
      } else {
        __builtin_amdgcn_s_wait_tensorcnt((short)0);
      }
    }
    __syncthreads();  // current tile visible to all waves

    fa_tile(st, &Kbuf[cur][0], &Vbuf[cur][0], Bias, &Plds[wave * 16 * STR],
            qa0, qa1, vone, ln, hi);

    __syncthreads();  // all waves done before next DMA overwrites the other buffer
  }
  fa_epilogue(st, O, b, qt, wave, ln, hi);
}
#endif  // HAVE_TDM

// ======== fallback: self-contained kernel (no workspace needed) ========
__global__ __launch_bounds__(kThreads) void fa64_fallback(
    const float* __restrict__ Q, const float* __restrict__ K,
    const float* __restrict__ V, const int* __restrict__ M,
    float* __restrict__ O) {
  __shared__ __align__(16) _Float16 Klds[kBK * STR];
  __shared__ __align__(16) _Float16 Vlds[kD * STR];
  __shared__ __align__(16) _Float16 Qlds[kBQ * STR];
  __shared__ __align__(16) _Float16 Plds[kNW * 16 * STR];
  __shared__ float Bias[kBK];

  const int tid  = threadIdx.x;
  const int wave = tid >> 5;
  const int lane = tid & 31;
  const int ln   = lane & 15;
  const int hi   = lane >> 4;
  const int b    = blockIdx.y;
  const int qt   = blockIdx.x * kBQ;

  const float* Qg = Q + ((size_t)b * kLq + qt) * kD;
  const float* Kg = K + (size_t)b * kLk * kD;
  const float* Vg = V + (size_t)b * kLk * kD;
  const int*   Mg = M + (size_t)b * kLk;

#pragma unroll
  for (int i = 0; i < (kBQ * kD / 4) / kThreads; ++i) {
    int idx = tid + i * kThreads;
    int row = idx >> 4;
    int c4  = idx & 15;
    v4f d = *(const v4f*)(Qg + (size_t)row * kD + c4 * 4);
    v4h h;
#pragma unroll
    for (int j = 0; j < 4; ++j) h[j] = (_Float16)(d[j] * kQScale);
    *(v4h*)(&Qlds[row * STR + c4 * 4]) = h;
  }
  __syncthreads();
  const v16h qa0 = ld8x2(&Qlds[(wave * 16 + ln) * STR +  0 + hi * 8]);
  const v16h qa1 = ld8x2(&Qlds[(wave * 16 + ln) * STR + 32 + hi * 8]);
  const v16h vone = make_vone(ln);

  FaState st;
  fa_init(st);

  for (int kb = 0; kb < kLk; kb += kBK) {
    __syncthreads();
#pragma unroll
    for (int i = 0; i < (kBK * kD / 4) / kThreads; ++i) {
      int idx = tid + i * kThreads;
      int row = idx >> 4;
      int c4  = idx & 15;
      v4f dk = *(const v4f*)(Kg + (size_t)(kb + row) * kD + c4 * 4);
      v4h hk;
#pragma unroll
      for (int j = 0; j < 4; ++j) hk[j] = (_Float16)dk[j];
      *(v4h*)(&Klds[row * STR + c4 * 4]) = hk;
    }
#pragma unroll
    for (int i = 0; i < (kBK * kD / 8) / kThreads; ++i) {
      int idx = tid + i * kThreads;
      int p   = idx >> 4;
      int c4  = idx & 15;
      v4f a0 = *(const v4f*)(Vg + (size_t)(kb + 2 * p    ) * kD + c4 * 4);
      v4f a1 = *(const v4f*)(Vg + (size_t)(kb + 2 * p + 1) * kD + c4 * 4);
#pragma unroll
      for (int j = 0; j < 4; ++j) {
        h2 t = {(_Float16)a0[j], (_Float16)a1[j]};
        *(h2*)(&Vlds[(c4 * 4 + j) * STR + 2 * p]) = t;
      }
    }
    if (tid < kBK) Bias[tid] = Mg[kb + tid] ? 0.0f : -INFINITY;
    __syncthreads();

    fa_tile(st, Klds, Vlds, Bias, &Plds[wave * 16 * STR], qa0, qa1, vone, ln, hi);
  }
  fa_epilogue(st, O, b, qt, wave, ln, hi);
}

extern "C" void kernel_launch(void* const* d_in, const int* in_sizes, int n_in,
                              void* d_out, int out_size, void* d_ws, size_t ws_size,
                              hipStream_t stream) {
  (void)in_sizes; (void)n_in; (void)out_size;
  const float* q = (const float*)d_in[0];
  const float* k = (const float*)d_in[1];
  const float* v = (const float*)d_in[2];
  const int*   m = (const int*)d_in[3];
  float* o = (float*)d_out;

  const size_t qbytes = (size_t)kB * kLq * kD * sizeof(_Float16);  // 2 MB
  const size_t kbytes = (size_t)kB * kLk * kD * sizeof(_Float16);  // 2 MB
  const size_t vbytes = kbytes;                                    // 2 MB
  const size_t bbytes = (size_t)kB * kLk * sizeof(float);          // 64 KB
  const size_t need   = qbytes + kbytes + vbytes + bbytes;

#ifdef HAVE_TDM
  if (ws_size >= need) {
    _Float16* qh = (_Float16*)d_ws;
    _Float16* kh = (_Float16*)((char*)d_ws + qbytes);
    _Float16* vt = (_Float16*)((char*)d_ws + qbytes + kbytes);
    float*    bw = (float*)((char*)d_ws + qbytes + kbytes + vbytes);
    const int tot = kB * kLq * kD / 4 + kB * kLk * kD / 4 + kB * kLk;
    prep_cvt<<<(tot + 255) / 256, 256, 0, stream>>>(q, k, m, qh, kh, bw);
    prep_vt<<<dim3(kLk / 64, kB), 256, 0, stream>>>(v, vt);
    fa64_tdm<<<dim3(kLq / kBQ, kB), kThreads, 0, stream>>>(qh, kh, vt, bw, o);
    return;
  }
#endif
  (void)ws_size; (void)d_ws;
  fa64_fallback<<<dim3(kLq / kBQ, kB), kThreads, 0, stream>>>(q, k, v, m, o);
}
